// GATHeadingPredictor_18545668784844
// MI455X (gfx1250) — compile-verified, matrix operand order checked by reading, not run on hardware
//
#include <hip/hip_runtime.h>
#include <hip/hip_bf16.h>

typedef __attribute__((ext_vector_type(16))) _Float16 v16h;
typedef __attribute__((ext_vector_type(8)))  _Float16 v8h;
typedef __attribute__((ext_vector_type(8)))  float    v8f;

#define NEG_SLOPE 0.2f
#define BN_EPS    1e-5f
#define CDIV(a, b) (((a) + (b) - 1) / (b))

// ---------------------------------------------------------------------------
// Order-preserving float<->uint map so segment-max can use atomicMax(u32).
// f>=0: set sign bit; f<0: bitwise-not. Monotone, and 0 is below every mapped
// finite value (mapped(-inf)=0x007FFFFF), so memset(0) is a valid identity
// because every dst has a self-loop (non-empty segment).
// ---------------------------------------------------------------------------
__device__ __forceinline__ unsigned mapf(float f) {
  unsigned b = __float_as_uint(f);
  return (b & 0x80000000u) ? ~b : (b | 0x80000000u);
}
__device__ __forceinline__ float unmapf(unsigned u) {
  unsigned b = (u & 0x80000000u) ? (u & 0x7fffffffu) : ~u;
  return __uint_as_float(b);
}

// ---------------------------------------------------------------------------
// Pack a row-major f32 weight [Ksrc x Nsrc] into f16 WMMA B-fragments,
// zero-padded to [Kpad x Npad]. Layout: chunk ((nt*KT + t)*32 + lane)*8 + i
// holds W[t*16 + (lane/16)*8 + i][nt*16 + lane%16]  (16x16 K-subtile t).
// ---------------------------------------------------------------------------
__global__ void pack_b_kernel(const float* __restrict__ W, _Float16* __restrict__ Bp,
                              int Ksrc, int Kpad, int Nsrc, int Npad) {
  int idx = blockIdx.x * blockDim.x + threadIdx.x;
  int total = Kpad * Npad;
  if (idx >= total) return;
  int i    = idx & 7;
  int lane = (idx >> 3) & 31;
  int KT   = Kpad >> 4;
  int t    = (idx >> 8) % KT;
  int nt   = (idx >> 8) / KT;
  int k = t * 16 + ((lane >> 4) << 3) + i;
  int n = nt * 16 + (lane & 15);
  float v = (k < Ksrc && n < Nsrc) ? W[k * Nsrc + n] : 0.0f;
  Bp[idx] = (_Float16)v;
}

// x [N,12] f32 -> xf16 [N,32] f16 zero-padded (K padded to a WMMA step).
__global__ void pad_x_kernel(const float* __restrict__ x, _Float16* __restrict__ xf, int N) {
  int idx = blockIdx.x * blockDim.x + threadIdx.x;
  if (idx >= N * 32) return;
  int n = idx >> 5, c = idx & 31;
  xf[idx] = (_Float16)(c < 12 ? x[n * 12 + c] : 0.0f);
}

// ---------------------------------------------------------------------------
// WMMA GEMM: out[Mx?] = A(f16 row-major, ld=K) x Bp(packed f16) [+bias][relu]
// One wave per 16x16 tile; blockDim=(32, Ncols/16); gridDim.x = M/16.
// MODE 0: raw f32 store           MODE 1: bias+relu -> f32 AND f16
// MODE 2: bias+relu -> f16 only   MODE 3: bias -> f32, cols < nvalid only
// ---------------------------------------------------------------------------
template <int MODE>
__global__ void gemm_wmma_kernel(const _Float16* __restrict__ A,
                                 const _Float16* __restrict__ Bp,
                                 const float* __restrict__ bias,
                                 float* __restrict__ outf, _Float16* __restrict__ outh,
                                 int K, int ldo, int nvalid) {
  const int lane = threadIdx.x;        // 0..31
  const int ct   = threadIdx.y;        // column tile
  const int rt   = blockIdx.x;         // row tile
  const int KT   = K >> 4;
  const int row0 = rt * 16 + (lane & 15);
  const int kb   = (lane >> 4) << 3;   // 0 or 8
  const _Float16* arow = A + (size_t)row0 * K;
  const _Float16* bp   = Bp + (size_t)ct * KT * 256;

  v8f acc = {};
  for (int s = 0; s < (K >> 5); ++s) {
    v8h a0 = *(const v8h*)(arow + s * 32 + kb);
    v8h a1 = *(const v8h*)(arow + s * 32 + 16 + kb);
    v8h b0 = *(const v8h*)(bp + (size_t)(2 * s) * 256 + lane * 8);
    v8h b1 = *(const v8h*)(bp + (size_t)(2 * s + 1) * 256 + lane * 8);
    v16h a, b;
#pragma unroll
    for (int i = 0; i < 8; ++i) {
      a[i] = a0[i]; a[i + 8] = a1[i];
      b[i] = b0[i]; b[i + 8] = b1[i];
    }
    acc = __builtin_amdgcn_wmma_f32_16x16x32_f16(false, a, false, b, (short)0, acc,
                                                 false, false);
  }

  // C layout: vgpr r -> M = (lane/16)*8 + r, N = lane%16
  const int mbase = rt * 16 + ((lane >> 4) << 3);
  const int ncol  = ct * 16 + (lane & 15);
#pragma unroll
  for (int r = 0; r < 8; ++r) {
    float v  = acc[r];
    int  row = mbase + r;
    if (MODE == 0) {
      outf[(size_t)row * ldo + ncol] = v;
    } else if (MODE == 1) {
      v += bias[ncol]; v = v > 0.f ? v : 0.f;
      outf[(size_t)row * ldo + ncol] = v;
      outh[(size_t)row * ldo + ncol] = (_Float16)v;
    } else if (MODE == 2) {
      v += bias[ncol]; v = v > 0.f ? v : 0.f;
      outh[(size_t)row * ldo + ncol] = (_Float16)v;
    } else {
      if (ncol < nvalid) {
        v += bias[ncol];
        outf[(size_t)row * ldo + ncol] = v;
      }
    }
  }
}

// Per-(node,head) attention projections: as = <xl[n,h,:],a_src[h]>, ad likewise.
__global__ void node_attn_kernel(const float* __restrict__ xl,
                                 const float* __restrict__ asrc,
                                 const float* __restrict__ adst,
                                 float* __restrict__ as_, float* __restrict__ ad_,
                                 int N, int heads, int ch) {
  int idx = blockIdx.x * blockDim.x + threadIdx.x;
  if (idx >= N * heads) return;
  int n = idx / heads, h = idx % heads;
  const float* xr = xl + (size_t)n * 128 + h * ch;
  float s1 = 0.f, s2 = 0.f;
  for (int c = 0; c < ch; ++c) {
    float v = xr[c];
    s1 += v * asrc[h * ch + c];
    s2 += v * adst[h * ch + c];
  }
  as_[idx] = s1;
  ad_[idx] = s2;
}

// Column sums of edge_attr [E,4] for the self-loop 'mean' fill value.
__global__ void edge_mean_kernel(const float* __restrict__ ea, float* msum, int E) {
  int tid = blockIdx.x * blockDim.x + threadIdx.x;
  int stride = gridDim.x * blockDim.x;
  float s0 = 0.f, s1 = 0.f, s2 = 0.f, s3 = 0.f;
  for (int e = tid; e < E; e += stride) {
    s0 += ea[e * 4 + 0]; s1 += ea[e * 4 + 1];
    s2 += ea[e * 4 + 2]; s3 += ea[e * 4 + 3];
  }
  atomicAdd(&msum[0], s0); atomicAdd(&msum[1], s1);
  atomicAdd(&msum[2], s2); atomicAdd(&msum[3], s3);
}

// P[d,h] = <We[d, h*32:(h+1)*32], aedge1[h]>  (folds ea@We . aedge into 4x4)
__global__ void compute_P_kernel(const float* __restrict__ We,
                                 const float* __restrict__ aedge, float* P) {
  int d = threadIdx.x >> 2, h = threadIdx.x & 3;   // 16 threads
  float s = 0.f;
  for (int c = 0; c < 32; ++c) s += We[d * 128 + h * 32 + c] * aedge[h * 32 + c];
  P[d * 4 + h] = s;
}

// alpha = leaky_relu(as[src]+ad[dst](+edge msg)); segment-max via mapped atomicMax.
__global__ void edge_alpha_kernel(const long long* __restrict__ ei, int E, int N,
                                  const float* __restrict__ as_, const float* __restrict__ ad_,
                                  const float* __restrict__ ea, const float* __restrict__ msum,
                                  const float* __restrict__ P, int hasmsg, int heads,
                                  float* __restrict__ alpha, unsigned* __restrict__ amaxU) {
  int idx = blockIdx.x * blockDim.x + threadIdx.x;
  int tot = (E + N) * heads;
  if (idx >= tot) return;
  int e = idx / heads, h = idx % heads;
  int src, dst;
  if (e < E) { src = (int)ei[e]; dst = (int)ei[E + e]; }
  else       { src = dst = e - E; }
  float a = as_[src * heads + h] + ad_[dst * heads + h];
  if (hasmsg) {
    float invE = 1.0f / (float)E;
#pragma unroll
    for (int d = 0; d < 4; ++d) {
      float v = (e < E) ? ea[e * 4 + d] : msum[d] * invE;
      a += v * P[d * heads + h];
    }
  }
  a = a > 0.f ? a : NEG_SLOPE * a;
  alpha[idx] = a;
  atomicMax(&amaxU[dst * heads + h], mapf(a));
}

// ex = exp(alpha - amax[dst]); denom[dst] += ex  (alpha overwritten in place).
__global__ void edge_exp_kernel(const long long* __restrict__ ei, int E, int N, int heads,
                                const unsigned* __restrict__ amaxU,
                                float* __restrict__ alpha, float* __restrict__ denom) {
  int idx = blockIdx.x * blockDim.x + threadIdx.x;
  int tot = (E + N) * heads;
  if (idx >= tot) return;
  int e = idx / heads, h = idx % heads;
  int dst = (e < E) ? (int)ei[E + e] : (e - E);
  float ex = __expf(alpha[idx] - unmapf(amaxU[dst * heads + h]));
  alpha[idx] = ex;
  atomicAdd(&denom[dst * heads + h], ex);
}

// scat[dst,:] += coef * xl[src,:]; one thread per (edge, 32-channel chunk).
__global__ void scatter_kernel(const long long* __restrict__ ei, int E, int N, int heads,
                               const float* __restrict__ ex, const float* __restrict__ denom,
                               const float* __restrict__ xl, float* __restrict__ scat) {
  int idx = blockIdx.x * blockDim.x + threadIdx.x;
  int tot = (E + N) * 4;
  if (idx >= tot) return;
  int e = idx >> 2, q = idx & 3;
  int src, dst;
  if (e < E) { src = (int)ei[e]; dst = (int)ei[E + e]; }
  else       { src = dst = e - E; }
  int h = (heads == 4) ? q : 0;
  float coef = ex[e * heads + h] / (denom[dst * heads + h] + 1e-16f);
  const float* xr = xl + (size_t)src * 128 + q * 32;
  float* orow     = scat + (size_t)dst * 128 + q * 32;
#pragma unroll
  for (int c = 0; c < 32; ++c) atomicAdd(&orow[c], xr[c] * coef);
}

// Epilogue: z = relu(BN(scat + bias)) [+ h_prev]; writes f32 and f16 copies.
__global__ void post_kernel(const float* __restrict__ scat, const float* __restrict__ bias,
                            const float* __restrict__ g, const float* __restrict__ bb,
                            const float* __restrict__ m, const float* __restrict__ v,
                            const float* __restrict__ hprev, int hasres,
                            float* __restrict__ h, _Float16* __restrict__ hf, int N) {
  int idx = blockIdx.x * blockDim.x + threadIdx.x;
  if (idx >= N * 128) return;
  int c = idx & 127;
  float z = scat[idx] + bias[c];
  z = (z - m[c]) * g[c] * rsqrtf(v[c] + BN_EPS) + bb[c];
  z = z > 0.f ? z : 0.f;
  if (hasres) z += hprev[idx];
  h[idx]  = z;
  hf[idx] = (_Float16)z;
}

extern "C" void kernel_launch(void* const* d_in, const int* in_sizes, int n_in,
                              void* d_out, int out_size, void* d_ws, size_t ws_size,
                              hipStream_t stream) {
  const float*     x      = (const float*)d_in[0];
  const long long* ei     = (const long long*)d_in[1];   // int64 in reference
  const float*     ea     = (const float*)d_in[2];
  const float*     W0     = (const float*)d_in[3];
  const float*     b0     = (const float*)d_in[4];
  const float*     W1     = (const float*)d_in[5];
  const float*     asrc1  = (const float*)d_in[6];
  const float*     adst1  = (const float*)d_in[7];
  const float*     We     = (const float*)d_in[8];
  const float*     aedge1 = (const float*)d_in[9];
  const float*     bias1  = (const float*)d_in[10];
  const float*     W2     = (const float*)d_in[11];
  const float*     asrc2  = (const float*)d_in[12];
  const float*     adst2  = (const float*)d_in[13];
  const float*     bias2  = (const float*)d_in[14];
  const float*     W3     = (const float*)d_in[15];
  const float*     asrc3  = (const float*)d_in[16];
  const float*     adst3  = (const float*)d_in[17];
  const float*     bias3  = (const float*)d_in[18];
  const float*     bn_g   = (const float*)d_in[19];
  const float*     bn_b   = (const float*)d_in[20];
  const float*     bn_m   = (const float*)d_in[21];
  const float*     bn_v   = (const float*)d_in[22];
  const float*     Wc1    = (const float*)d_in[23];
  const float*     bc1    = (const float*)d_in[24];
  const float*     Wc2    = (const float*)d_in[25];
  const float*     bc2    = (const float*)d_in[26];

  const int N    = in_sizes[0] / 12;
  const int E    = in_sizes[1] / 2;
  const int Etot = E + N;
  const int RT   = N / 16;               // N = 50000 = 3125*16

  // ---- carve workspace -----------------------------------------------------
  char*  wsb = (char*)d_ws;
  size_t off = 0;
  auto alloc = [&](size_t bytes) -> void* {
    void* p = wsb + off;
    off = (off + bytes + 255) & ~(size_t)255;
    return p;
  };
  _Float16* xf16  = (_Float16*)alloc((size_t)N * 32 * 2);
  _Float16* hf16  = (_Float16*)alloc((size_t)N * 128 * 2);
  float*    h     = (float*)   alloc((size_t)N * 128 * 4);
  float*    xl    = (float*)   alloc((size_t)N * 128 * 4);
  float*    scat  = (float*)   alloc((size_t)N * 128 * 4);
  _Float16* z1f16 = (_Float16*)alloc((size_t)N * 64 * 2);
  float*    as_   = (float*)   alloc((size_t)N * 4 * 4);
  float*    ad_   = (float*)   alloc((size_t)N * 4 * 4);
  float*    alp   = (float*)   alloc((size_t)Etot * 4 * 4);
  unsigned* amaxU = (unsigned*)alloc((size_t)N * 4 * 4);
  float*    denom = (float*)   alloc((size_t)N * 4 * 4);
  float*    msum  = (float*)   alloc(64);
  float*    P     = (float*)   alloc(64);
  _Float16* W0p   = (_Float16*)alloc(32 * 128 * 2);
  _Float16* W1p   = (_Float16*)alloc(128 * 128 * 2);
  _Float16* W2p   = (_Float16*)alloc(128 * 128 * 2);
  _Float16* W3p   = (_Float16*)alloc(128 * 128 * 2);
  _Float16* Wc1p  = (_Float16*)alloc(128 * 64 * 2);
  _Float16* Wc2p  = (_Float16*)alloc(64 * 80 * 2);

  // ---- weight packing + input conversion ----------------------------------
  pack_b_kernel<<<CDIV(32 * 128, 256), 256, 0, stream>>>(W0, W0p, 12, 32, 128, 128);
  pack_b_kernel<<<CDIV(128 * 128, 256), 256, 0, stream>>>(W1, W1p, 128, 128, 128, 128);
  pack_b_kernel<<<CDIV(128 * 128, 256), 256, 0, stream>>>(W2, W2p, 128, 128, 128, 128);
  pack_b_kernel<<<CDIV(128 * 128, 256), 256, 0, stream>>>(W3, W3p, 128, 128, 128, 128);
  pack_b_kernel<<<CDIV(128 * 64, 256), 256, 0, stream>>>(Wc1, Wc1p, 128, 128, 64, 64);
  pack_b_kernel<<<CDIV(64 * 80, 256), 256, 0, stream>>>(Wc2, Wc2p, 64, 64, 72, 80);
  pad_x_kernel<<<CDIV(N * 32, 256), 256, 0, stream>>>(x, xf16, N);

  // ---- encoder: h = relu(x@W0 + b0) ---------------------------------------
  gemm_wmma_kernel<1><<<RT, dim3(32, 8), 0, stream>>>(xf16, W0p, b0, h, hf16, 32, 128, 0);

  // ---- edge message precompute (layer 1 only) -----------------------------
  hipMemsetAsync(msum, 0, 4 * sizeof(float), stream);
  edge_mean_kernel<<<256, 256, 0, stream>>>(ea, msum, E);
  compute_P_kernel<<<1, 16, 0, stream>>>(We, aedge1, P);

  // ---- three GAT layers ---------------------------------------------------
  const float* Wp_l[3]    = {nullptr, nullptr, nullptr};
  const _Float16* Wpk[3]  = {W1p, W2p, W3p};
  const float* asrcL[3]   = {asrc1, asrc2, asrc3};
  const float* adstL[3]   = {adst1, adst2, adst3};
  const float* biasL[3]   = {bias1, bias2, bias3};
  const int    headsL[3]  = {4, 4, 1};
  (void)Wp_l;

  for (int L = 0; L < 3; ++L) {
    const int heads = headsL[L];
    const int ch    = 128 / heads;

    gemm_wmma_kernel<0><<<RT, dim3(32, 8), 0, stream>>>(hf16, Wpk[L], nullptr, xl,
                                                        nullptr, 128, 128, 0);
    node_attn_kernel<<<CDIV(N * heads, 256), 256, 0, stream>>>(xl, asrcL[L], adstL[L],
                                                               as_, ad_, N, heads, ch);
    hipMemsetAsync(amaxU, 0, (size_t)N * heads * 4, stream);
    hipMemsetAsync(denom, 0, (size_t)N * heads * 4, stream);
    hipMemsetAsync(scat, 0, (size_t)N * 128 * 4, stream);

    edge_alpha_kernel<<<CDIV(Etot * heads, 256), 256, 0, stream>>>(
        ei, E, N, as_, ad_, ea, msum, P, (L == 0) ? 1 : 0, heads, alp, amaxU);
    edge_exp_kernel<<<CDIV(Etot * heads, 256), 256, 0, stream>>>(ei, E, N, heads, amaxU,
                                                                 alp, denom);
    scatter_kernel<<<CDIV(Etot * 4, 256), 256, 0, stream>>>(ei, E, N, heads, alp, denom,
                                                            xl, scat);
    post_kernel<<<CDIV(N * 128, 256), 256, 0, stream>>>(
        scat, biasL[L], bn_g + 128 * L, bn_b + 128 * L, bn_m + 128 * L, bn_v + 128 * L,
        h, (L > 0) ? 1 : 0, h, hf16, N);
  }

  // ---- MLP head: out = relu(h@Wc1+bc1)@Wc2 + bc2 --------------------------
  gemm_wmma_kernel<2><<<RT, dim3(32, 4), 0, stream>>>(hf16, Wc1p, bc1, nullptr, z1f16,
                                                      128, 64, 0);
  gemm_wmma_kernel<3><<<RT, dim3(32, 5), 0, stream>>>(z1f16, Wc2p, bc2, (float*)d_out,
                                                      nullptr, 64, 72, 72);
}
